// AIMNet2_75514114998579
// MI455X (gfx1250) — matrix operand order repaired and checked by reading, not compiled, not run on hardware
//
#include <hip/hip_runtime.h>
#include <hip/hip_bf16.h>
#include <math.h>

// ---------------- problem constants ----------------
static constexpr int kB  = 16;
static constexpr int kN  = 256;
static constexpr int kM  = 64;
static constexpr int kC  = 128;
static constexpr int kS  = 16;
static constexpr int kK  = 12;
static constexpr int kBN = kB * kN;          // 4096 atoms
static constexpr int NIN_A  = 3712;          // C + C*(S+K)
static constexpr int NIN_Q  = 3741;          // NIN_A + 1 + (S+K)
static constexpr int KP_A   = 3712;          // pad32(3712)
static constexpr int KP_AQ  = 3744;          // pad32(3741)
static constexpr int KP_CAT = 4000;          // pad32(256+3741)
static constexpr int OUTW   = 769;           // 1 + 256 + 256 + 256

typedef __attribute__((ext_vector_type(16))) __bf16 v16bf;
typedef __attribute__((ext_vector_type(8)))  float  v8f;
typedef int vi4 __attribute__((vector_size(16)));    // matches builtin's V4i
union Frag { uint4 q[2]; v16bf v; };

// -------- CDNA5 async global->LDS path (ASYNCcnt), guarded for portability ----
#if defined(__has_builtin)
#if __has_builtin(__builtin_amdgcn_global_load_async_to_lds_b128) && \
    __has_builtin(__builtin_amdgcn_s_wait_asynccnt)
#define USE_ASYNC_LDS 1
#endif
#endif
#ifndef USE_ASYNC_LDS
#define USE_ASYNC_LDS 0
#endif

__device__ __forceinline__ void async_copy_b128(const unsigned short* g,
                                                unsigned short* l) {
#if USE_ASYNC_LDS
  __builtin_amdgcn_global_load_async_to_lds_b128(
      (__attribute__((address_space(1))) vi4*)const_cast<unsigned short*>(g),
      (__attribute__((address_space(3))) vi4*)l, 0, 0);
#else
  *(uint4*)l = *(const uint4*)g;
#endif
}
__device__ __forceinline__ void async_wait_all() {
#if USE_ASYNC_LDS
  __builtin_amdgcn_s_wait_asynccnt(0);
#endif
}

__device__ __forceinline__ unsigned short f2bf(float f) {
  unsigned int u = __float_as_uint(f);
  u += 0x7FFFu + ((u >> 16) & 1u);              // round-to-nearest-even
  return (unsigned short)(u >> 16);
}
__device__ __forceinline__ float bf2f(unsigned short h) {
  return __uint_as_float(((unsigned int)h) << 16);
}
__device__ __forceinline__ float gelu_f(float x) {
  return 0.5f * x * (1.0f + erff(x * 0.70710678118654752440f));
}
__device__ __forceinline__ v8f wmma_bf16(const Frag& a, const Frag& b, v8f c) {
  return __builtin_amdgcn_wmma_f32_16x16x32_bf16(false, a.v, false, b.v,
                                                 (short)0, c, false, false);
}

// ---------------- weight prep: f32 -> padded bf16 ----------------
__global__ void k_prep_w(const float* __restrict__ W, const float* __restrict__ b,
                         unsigned short* __restrict__ Wb, float* __restrict__ bb,
                         int nin, int nout, int KP, int NP) {
  int idx = blockIdx.x * 256 + threadIdx.x;
  int tot = KP * NP;
  if (idx < tot) {
    int k = idx / NP, n = idx % NP;
    float v = (k < nin && n < nout) ? W[(size_t)k * nout + n] : 0.0f;
    Wb[idx] = f2bf(v);
  }
  if (idx < NP) bb[idx] = (idx < nout) ? b[idx] : 0.0f;
}

// ---------------- atomic-feature embedding ----------------
__global__ void k_embed(const float* __restrict__ afv, const int* __restrict__ numbers,
                        float* __restrict__ a) {
  int i = blockIdx.x * 256 + threadIdx.x;
  if (i >= kBN * kC) return;
  int bn = i >> 7, c = i & 127;
  a[i] = afv[numbers[bn] * kC + c];
}

// ---------------- geometry: per-pair radial basis + vector basis ----------------
// cols 0..15 = gs[s], cols 16+d*12+k = u[d]*(sum_s gs[s]*comb[s][k]), cols 52..63 = 0
__global__ void k_geom(const float* __restrict__ coord, const int* __restrict__ nbmat,
                       const float* __restrict__ comb_a, const float* __restrict__ comb_q,
                       unsigned short* __restrict__ Bpa, unsigned short* __restrict__ Bpq) {
  int pair = blockIdx.x * 256 + threadIdx.x;
  if (pair >= kBN * kM) return;
  int bn = pair >> 6;
  int b  = bn >> 8, n = bn & 255;
  int nb = nbmat[pair];
  const float* ci = coord + (size_t)bn * 3;
  const float* cj = coord + ((size_t)(b * kN + nb)) * 3;
  float rx = cj[0] - ci[0], ry = cj[1] - ci[1], rz = cj[2] - ci[2];
  float d = sqrtf(rx * rx + ry * ry + rz * rz + 1e-12f);
  float fc = 0.0f;
  if (d < 5.0f) fc = 0.5f * cosf(0.62831853071795864769f * d) + 0.5f;  // pi/RC
  if (nb == n) fc = 0.0f;
  float inv = 1.0f / d;
  float ux = rx * inv, uy = ry * inv, uz = rz * inv;
  float gs[kS];
#pragma unroll
  for (int s = 0; s < kS; ++s) {
    float sh = 0.8f + 0.28f * (float)s;        // linspace(0.8, 5.0, 16)
    float t = d - sh;
    gs[s] = expf(-4.0f * t * t) * fc;
  }
  unsigned short* pa = Bpa + (size_t)pair * 64;
  unsigned short* pq = Bpq + (size_t)pair * 64;
#pragma unroll
  for (int s = 0; s < kS; ++s) { unsigned short g = f2bf(gs[s]); pa[s] = g; pq[s] = g; }
#pragma unroll
  for (int k = 0; k < kK; ++k) {
    float ta = 0.f, tq = 0.f;
#pragma unroll
    for (int s = 0; s < kS; ++s) {
      ta += gs[s] * comb_a[s * kK + k];
      tq += gs[s] * comb_q[s * kK + k];
    }
    pa[16 + k] = f2bf(ux * ta); pa[28 + k] = f2bf(uy * ta); pa[40 + k] = f2bf(uz * ta);
    pq[16 + k] = f2bf(ux * tq); pq[28 + k] = f2bf(uy * tq); pq[40 + k] = f2bf(uz * tq);
  }
#pragma unroll
  for (int j = 52; j < 64; ++j) { pa[j] = 0; pq[j] = 0; }
}

// ---------------- in_a convolution: per-atom 128x64 @ 64x52 via WMMA ----------------
__global__ __launch_bounds__(128) void k_conv_a(
    const float* __restrict__ a_cur, const int* __restrict__ nbmat,
    const unsigned short* __restrict__ Bpa, unsigned short* __restrict__ X, int ldX) {
  __shared__ int nbL[kM];
  __shared__ __align__(16) unsigned short ajT[kC][kM];   // A^T: [c][m]
  __shared__ __align__(16) unsigned short BpT[kM][kM];   // B^T: [col][m]
  __shared__ float OutS[kC][kM];
  int bn = blockIdx.x;
  int b  = bn >> 8;
  int t  = threadIdx.x;                                   // 0..127
  if (t < kM) nbL[t] = nbmat[(size_t)bn * kM + t];
  __syncthreads();
  { // gather neighbor features transposed (c-major)
    int c = t;
    for (int m = 0; m < kM; ++m)
      ajT[c][m] = f2bf(a_cur[((size_t)(b * kN + nbL[m])) * kC + c]);
  }
  { // transpose per-pair B matrix into LDS
    const unsigned short* Bp = Bpa + (size_t)bn * (kM * 64);
    for (int i = 0; i < 32; ++i) {
      int e = t * 32 + i;
      int m = e >> 6, col = e & 63;
      BpT[col][m] = Bp[m * 64 + col];
    }
  }
  __syncthreads();
  int w = t >> 5, lane = t & 31;
  int h = lane >> 4, l16 = lane & 15;
  v8f zero = {0.f, 0.f, 0.f, 0.f, 0.f, 0.f, 0.f, 0.f};
  v8f acc[2][4];
#pragma unroll
  for (int i = 0; i < 2; ++i)
#pragma unroll
    for (int j = 0; j < 4; ++j) acc[i][j] = zero;
#pragma unroll
  for (int k0 = 0; k0 < 64; k0 += 32) {
    Frag A[2], Bf[4];
#pragma unroll
    for (int i = 0; i < 2; ++i) {
      int r = (w * 2 + i) * 16 + l16;
      A[i].q[0] = *(const uint4*)&ajT[r][k0 + h * 16];
      A[i].q[1] = *(const uint4*)&ajT[r][k0 + h * 16 + 8];
    }
#pragma unroll
    for (int j = 0; j < 4; ++j) {
      int nn = j * 16 + l16;
      Bf[j].q[0] = *(const uint4*)&BpT[nn][k0 + h * 16];
      Bf[j].q[1] = *(const uint4*)&BpT[nn][k0 + h * 16 + 8];
    }
#pragma unroll
    for (int i = 0; i < 2; ++i)
#pragma unroll
      for (int j = 0; j < 4; ++j)
        acc[i][j] = wmma_bf16(A[i], Bf[j], acc[i][j]);
  }
#pragma unroll
  for (int i = 0; i < 2; ++i)
#pragma unroll
    for (int j = 0; j < 4; ++j)
#pragma unroll
      for (int v = 0; v < 8; ++v) {
        int row = (w * 2 + i) * 16 + v + h * 8;
        int col = j * 16 + l16;
        OutS[row][col] = acc[i][j][v];
      }
  __syncthreads();
  { // feature assembly: [a | s-conv | |v|-norms]
    int c = t;
    unsigned short* Xr = X + (size_t)bn * ldX;
    Xr[c] = f2bf(a_cur[(size_t)bn * kC + c]);
#pragma unroll
    for (int s = 0; s < kS; ++s) Xr[kC + c * kS + s] = f2bf(OutS[c][s]);
#pragma unroll
    for (int k = 0; k < kK; ++k) {
      float v0 = OutS[c][16 + k], v1 = OutS[c][28 + k], v2 = OutS[c][40 + k];
      Xr[kC + kC * kS + c * kK + k] =
          f2bf(sqrtf(v0 * v0 + v1 * v1 + v2 * v2 + 1e-8f));
    }
  }
}

// ---------------- in_q convolution (C == 1, VALU) ----------------
__global__ __launch_bounds__(64) void k_in_q(
    const float* __restrict__ q_cur, const int* __restrict__ nbmat,
    const unsigned short* __restrict__ Bpq, unsigned short* __restrict__ X, int ldX) {
  __shared__ float qj[kM];
  __shared__ float vals[52];
  int bn = blockIdx.x, t = threadIdx.x;
  int b = bn >> 8;
  int nb = nbmat[(size_t)bn * kM + t];
  qj[t] = q_cur[b * kN + nb];
  __syncthreads();
  if (t < 52) {
    const unsigned short* Bp = Bpq + (size_t)bn * (kM * 64);
    float acc = 0.f;
    for (int m = 0; m < kM; ++m) acc += qj[m] * bf2f(Bp[m * 64 + t]);
    vals[t] = acc;
  }
  __syncthreads();
  unsigned short* Xr = X + (size_t)bn * ldX + NIN_A;
  if (t == 0) Xr[0] = f2bf(q_cur[bn]);
  if (t < kS) Xr[1 + t] = f2bf(vals[t]);
  if (t < kK) {
    float v0 = vals[16 + t], v1 = vals[28 + t], v2 = vals[40 + t];
    Xr[1 + kS + t] = f2bf(sqrtf(v0 * v0 + v1 * v1 + v2 * v2 + 1e-8f));
  }
  if (t < 3) Xr[29 + t] = 0;                    // zero K-pad (cols 3741..3743)
}

// ---------------- generic bf16 GEMM + bias + optional exact GELU ----------------
// Y = act(X[KP] @ W[KP x NP] + bias); 128x64 block tile, 8 waves of 32x32.
// Double-buffered LDS; X tile staged via GLOBAL_LOAD_ASYNC_TO_LDS_B128 (ASYNCcnt),
// W tile staged via VGPR transpose; next-but-one tiles prefetched.
__global__ __launch_bounds__(256) void k_gemm(
    const unsigned short* __restrict__ X, int KP,
    const unsigned short* __restrict__ W, const float* __restrict__ bias,
    int NP, int act,
    float* __restrict__ Yf, int ldYf,
    unsigned short* __restrict__ Yb, int ldYb) {
  __shared__ __align__(16) unsigned short Xs[2][128][32];
  __shared__ __align__(16) unsigned short WsT[2][64][32];
  int t = threadIdx.x;
  int rBase = blockIdx.x * 128;
  int cBase = blockIdx.y * 64;
  int w = t >> 5, lane = t & 31;
  int wr = w & 3, wc = w >> 2;
  int h = lane >> 4, l16 = lane & 15;

  // per-thread staging coordinates (fixed for all k-steps)
  int e  = t * 8;
  int xr0 = e >> 5,          xc0 = e & 31;
  int xr1 = (e + 2048) >> 5, xc1 = (e + 2048) & 31;
  int wk = e >> 6, wn = e & 63;

  auto stage = [&](int k0, int buf) {
    // X tile: two async b128 copies per thread (128x32 bf16)
    async_copy_b128(&X[(size_t)(rBase + xr0) * KP + k0 + xc0], &Xs[buf][xr0][xc0]);
    async_copy_b128(&X[(size_t)(rBase + xr1) * KP + k0 + xc1], &Xs[buf][xr1][xc1]);
    // W tile: 32xNP-tile row-major load, transposed store to LDS
    uint4 wv = *(const uint4*)&W[(size_t)(k0 + wk) * NP + cBase + wn];
    const unsigned short* ws = reinterpret_cast<const unsigned short*>(&wv);
#pragma unroll
    for (int i = 0; i < 8; ++i) WsT[buf][wn + i][wk] = ws[i];
    // prefetch next-but-one tiles toward L2
    __builtin_prefetch(&X[(size_t)(rBase + xr0) * KP + k0 + 64 + xc0], 0, 0);
    __builtin_prefetch(&W[(size_t)(k0 + 64 + wk) * NP + cBase + wn], 0, 0);
  };

  v8f zero = {0.f, 0.f, 0.f, 0.f, 0.f, 0.f, 0.f, 0.f};
  v8f acc[2][2];
  acc[0][0] = zero; acc[0][1] = zero; acc[1][0] = zero; acc[1][1] = zero;

  int nk = KP >> 5;
  stage(0, 0);
  async_wait_all();
  __syncthreads();
  for (int ks = 0; ks < nk; ++ks) {
    int cur = ks & 1;
    if (ks + 1 < nk) stage((ks + 1) << 5, cur ^ 1);
    Frag A[2], Bf[2];
#pragma unroll
    for (int i = 0; i < 2; ++i) {
      int r = wr * 32 + i * 16 + l16;
      A[i].q[0] = *(const uint4*)&Xs[cur][r][h * 16];
      A[i].q[1] = *(const uint4*)&Xs[cur][r][h * 16 + 8];
    }
#pragma unroll
    for (int j = 0; j < 2; ++j) {
      int nn = wc * 32 + j * 16 + l16;
      Bf[j].q[0] = *(const uint4*)&WsT[cur][nn][h * 16];
      Bf[j].q[1] = *(const uint4*)&WsT[cur][nn][h * 16 + 8];
    }
#pragma unroll
    for (int i = 0; i < 2; ++i)
#pragma unroll
      for (int j = 0; j < 2; ++j)
        acc[i][j] = wmma_bf16(A[i], Bf[j], acc[i][j]);
    async_wait_all();
    __syncthreads();
  }
#pragma unroll
  for (int i = 0; i < 2; ++i)
#pragma unroll
    for (int j = 0; j < 2; ++j)
#pragma unroll
      for (int v = 0; v < 8; ++v) {
        int row = rBase + wr * 32 + i * 16 + v + h * 8;
        int col = cBase + wc * 32 + j * 16 + l16;
        float val = acc[i][j][v] + bias[col];
        if (act) val = gelu_f(val);
        if (Yf) Yf[(size_t)row * ldYf + col] = val;
        if (Yb) Yb[(size_t)row * ldYb + col] = f2bf(val);
      }
}

// ---------------- heads / NSE / residuals / glue ----------------
__global__ void k_head_scalar(const float* __restrict__ Y, int ldY,
                              const float* __restrict__ qprev,
                              float* __restrict__ qout, float* __restrict__ f2out) {
  int bn = blockIdx.x * 256 + threadIdx.x;
  if (bn >= kBN) return;
  float q0 = Y[(size_t)bn * ldY];
  if (qprev) q0 += qprev[bn];
  qout[bn] = q0;
  if (f2out) { float f = Y[(size_t)bn * ldY + 1]; f2out[bn] = f * f; }
}

__global__ void k_add_da(const float* __restrict__ Y, int ldY,
                         const float* __restrict__ a_in, float* __restrict__ a_out) {
  int i = blockIdx.x * 256 + threadIdx.x;
  if (i >= kBN * kC) return;
  int bn = i >> 7, c = i & 127;
  a_out[i] = a_in[i] + Y[(size_t)bn * ldY + 2 + c];
}

__global__ __launch_bounds__(256) void k_nse(const float* __restrict__ Qtot,
                                             const float* __restrict__ q,
                                             const float* __restrict__ f2,
                                             float* __restrict__ qout) {
  __shared__ float sq[256], sf[256];
  int b = blockIdx.x, n = threadIdx.x;
  int bn = b * kN + n;
  float qv = q[bn], fv = f2[bn];
  sq[n] = qv; sf[n] = fv;
  __syncthreads();
  for (int s = 128; s > 0; s >>= 1) {
    if (n < s) { sq[n] += sq[n + s]; sf[n] += sf[n + s]; }
    __syncthreads();
  }
  float dq = Qtot[b] - sq[0];
  float scale = dq / (sf[0] + 1e-6f);
  qout[bn] = qv + fv * scale;
}

__global__ void k_concat(const unsigned short* __restrict__ Xbase,
                         unsigned short* __restrict__ Xcat) {
  int i = blockIdx.x * 256 + threadIdx.x;
  if (i >= kBN * KP_AQ) return;
  int bn = i / KP_AQ, j = i % KP_AQ;
  Xcat[(size_t)bn * KP_CAT + 256 + j] =
      (j < NIN_Q) ? Xbase[(size_t)bn * KP_AQ + j] : (unsigned short)0;
}

__global__ void k_write_q(const float* __restrict__ q, float* __restrict__ out) {
  int bn = blockIdx.x * 256 + threadIdx.x;
  if (bn >= kBN) return;
  out[(size_t)bn * OUTW] = q[bn];
}

// ---------------- host orchestration ----------------
extern "C" void kernel_launch(void* const* d_in, const int* in_sizes, int n_in,
                              void* d_out, int out_size, void* d_ws, size_t ws_size,
                              hipStream_t stream) {
  (void)in_sizes; (void)out_size; (void)ws_size;
  const float* coord  = (const float*)d_in[0];
  const float* charge = (const float*)d_in[1];
  const float* afv    = (const float*)d_in[2];
  const float* comb_a = (const float*)d_in[3];
  const float* comb_q = (const float*)d_in[4];
  const int* numbers  = (const int*)d_in[n_in - 2];
  const int* nbmat    = (const int*)d_in[n_in - 1];
  float* out = (float*)d_out;

  // workspace carve-up (256B aligned)
  char* wp = (char*)d_ws;
  auto carve = [&](size_t bytes) -> void* {
    void* p = (void*)wp;
    wp += (bytes + 255) & ~(size_t)255;
    return p;
  };
  unsigned short* Bpa    = (unsigned short*)carve((size_t)kBN * kM * 64 * 2);
  unsigned short* Bpq    = (unsigned short*)carve((size_t)kBN * kM * 64 * 2);
  unsigned short* XA     = (unsigned short*)carve((size_t)kBN * KP_AQ * 2);
  unsigned short* Xbase  = (unsigned short*)carve((size_t)kBN * KP_AQ * 2);
  unsigned short* Xcat   = (unsigned short*)carve((size_t)kBN * KP_CAT * 2);
  unsigned short* H1     = (unsigned short*)carve((size_t)kBN * 384 * 2);
  unsigned short* H2     = (unsigned short*)carve((size_t)kBN * 256 * 2);
  float*          Yhead  = (float*)carve((size_t)kBN * 192 * 4);
  unsigned short* Warena = (unsigned short*)carve((size_t)16 * 1024 * 1024 * 2);
  float*          Barena = (float*)carve((size_t)16384 * 4);
  float* a_cur = (float*)carve((size_t)kBN * kC * 4);
  float* pa    = (float*)carve((size_t)kBN * kC * 4);
  float* q_cur = (float*)carve((size_t)kBN * 4);
  float* pq    = (float*)carve((size_t)kBN * 4);
  float* q_tmp = (float*)carve((size_t)kBN * 4);
  float* f2    = (float*)carve((size_t)kBN * 4);

  // layer table: 9 MLPs x 3 layers.
  // params pytree order: afv, comb_a, comb_q, emlps[0..5]{W,b}x3, mlps[0..2]{W,b}x3
  struct Layer { int nin, nout, KP, NP; size_t wOff, bOff; int wIdx, bIdx; };
  Layer lay[27];
  const int mlpBase[9] = {41, 47, 53,  5, 11, 17, 23, 29, 35};   // mlp0..2, e0..e5
  const int mlpNin[9]  = {3712, 3741, 3741, 3997, 3741, 3741, 3997, 3741, 3741};
  const int mlpNout[9] = {130, 130, 256, 130, 130, 256, 130, 130, 256};
  size_t wAcc = 0, bAcc = 0;
  for (int m = 0; m < 9; ++m)
    for (int l = 0; l < 3; ++l) {
      Layer& L = lay[m * 3 + l];
      L.nin  = (l == 0) ? mlpNin[m] : (l == 1 ? 384 : 256);
      L.nout = (l == 2) ? mlpNout[m] : (l == 0 ? 384 : 256);
      L.KP = (L.nin + 31) / 32 * 32;
      L.NP = (L.nout + 63) / 64 * 64;
      L.wIdx = mlpBase[m] + l * 2;
      L.bIdx = L.wIdx + 1;
      L.wOff = wAcc; L.bOff = bAcc;
      wAcc += (size_t)L.KP * L.NP; bAcc += L.NP;
    }

  for (int i = 0; i < 27; ++i) {
    Layer& L = lay[i];
    int tot = L.KP * L.NP;
    k_prep_w<<<(tot + 255) / 256, 256, 0, stream>>>(
        (const float*)d_in[L.wIdx], (const float*)d_in[L.bIdx],
        Warena + L.wOff, Barena + L.bOff, L.nin, L.nout, L.KP, L.NP);
  }

  k_embed<<<(kBN * kC) / 256, 256, 0, stream>>>(afv, numbers, a_cur);
  k_geom<<<(kBN * kM) / 256, 256, 0, stream>>>(coord, nbmat, comb_a, comb_q, Bpa, Bpq);

  auto gemm = [&](const unsigned short* Xp, int KPx, int li, int act,
                  float* Yf, int ldYf, unsigned short* Yb, int ldYb) {
    Layer& L = lay[li];
    dim3 g(kBN / 128, L.NP / 64);
    k_gemm<<<g, 256, 0, stream>>>(Xp, KPx, Warena + L.wOff, Barena + L.bOff,
                                  L.NP, act, Yf, ldYf, Yb, ldYb);
  };
  auto runMlp = [&](int mid, const unsigned short* Xin, int KPin, int actLast,
                    float* Yf, int ldYf, unsigned short* Yb, int ldYb) {
    gemm(Xin, KPin, mid * 3 + 0, 1, nullptr, 0, H1, 384);
    gemm(H1, 384, mid * 3 + 1, 1, nullptr, 0, H2, 256);
    gemm(H2, 256, mid * 3 + 2, actLast, Yf, ldYf, Yb, ldYb);
  };

  // ---- step 1: mlp0(in_a(a)), NSE
  k_conv_a<<<kBN, 128, 0, stream>>>(a_cur, nbmat, Bpa, XA, KP_A);
  runMlp(0, XA, KP_A, 0, Yhead, 192, nullptr, 0);
  k_head_scalar<<<kBN / 256, 256, 0, stream>>>(Yhead, 192, nullptr, q_tmp, f2);
  k_add_da<<<(kBN * kC) / 256, 256, 0, stream>>>(Yhead, 192, a_cur, a_cur);
  k_nse<<<kB, kN, 0, stream>>>(charge, q_tmp, f2, q_cur);

  // ---- step 2: mlp1([in_a|in_q]), NSE
  k_conv_a<<<kBN, 128, 0, stream>>>(a_cur, nbmat, Bpa, XA, KP_AQ);
  k_in_q<<<kBN, kM, 0, stream>>>(q_cur, nbmat, Bpq, XA, KP_AQ);
  runMlp(1, XA, KP_AQ, 1, Yhead, 192, nullptr, 0);
  k_head_scalar<<<kBN / 256, 256, 0, stream>>>(Yhead, 192, q_cur, q_tmp, f2);
  k_add_da<<<(kBN * kC) / 256, 256, 0, stream>>>(Yhead, 192, a_cur, a_cur);
  k_nse<<<kB, kN, 0, stream>>>(charge, q_tmp, f2, q_cur);

  // ---- base + aim (aim f32 -> d_out cols 1..256, bf16 -> Xcat cols 0..255)
  k_conv_a<<<kBN, 128, 0, stream>>>(a_cur, nbmat, Bpa, Xbase, KP_AQ);
  k_in_q<<<kBN, kM, 0, stream>>>(q_cur, nbmat, Bpq, Xbase, KP_AQ);
  runMlp(2, Xbase, KP_AQ, 1, out + 1, OUTW, Xcat, KP_CAT);
  k_concat<<<(kBN * KP_AQ + 255) / 256, 256, 0, stream>>>(Xbase, Xcat);

  // ---- two output passes
  for (int p = 0; p < 2; ++p) {
    runMlp(3 + 3 * p, Xcat, KP_CAT, 0, Yhead, 192, nullptr, 0);
    k_head_scalar<<<kBN / 256, 256, 0, stream>>>(Yhead, 192, q_cur, pq, nullptr);
    k_add_da<<<(kBN * kC) / 256, 256, 0, stream>>>(Yhead, 192, a_cur, pa);
    k_conv_a<<<kBN, 128, 0, stream>>>(pa, nbmat, Bpa, XA, KP_AQ);
    k_in_q<<<kBN, kM, 0, stream>>>(pq, nbmat, Bpq, XA, KP_AQ);
    runMlp(4 + 3 * p, XA, KP_AQ, 0, Yhead, 192, nullptr, 0);
    k_head_scalar<<<kBN / 256, 256, 0, stream>>>(Yhead, 192, pq, pq, nullptr);
    k_add_da<<<(kBN * kC) / 256, 256, 0, stream>>>(Yhead, 192, pa, pa);
    k_conv_a<<<kBN, 128, 0, stream>>>(pa, nbmat, Bpa, XA, KP_AQ);
    k_in_q<<<kBN, kM, 0, stream>>>(pq, nbmat, Bpq, XA, KP_AQ);
    runMlp(5 + 3 * p, XA, KP_AQ, 1, out + 257 + 256 * p, OUTW, nullptr, 0);
  }
  k_write_q<<<kBN / 256, 256, 0, stream>>>(q_cur, out);
}
